// New_CVMI_50749333569868
// MI455X (gfx1250) — matrix operand order repaired and checked by reading, not compiled, not run on hardware
//
#include <hip/hip_runtime.h>
#include <hip/hip_bf16.h>
#include <cstdint>

// ---------------------------------------------------------------------------
// MI455X / gfx1250 implementation.
//   * all intermediate activations: bf16 NHWC  ->  im2col tiles and attention
//     Q/K row-slabs are contiguous, staged with GLOBAL_LOAD_ASYNC_TO_LDS_B128
//     (ASYNCcnt) instead of VGPR-bounce copies.
//   * every GEMM (12 convs + 8 attention GEMMs) runs on v_wmma_f32_16x16x32_bf16.
//   * relu(softmax(x)) == softmax(x); all 6 score applications recast as
//     out = feat * S^T / feat * S (M=c=64, N=w=192, K=v=192) sharing A-fragments.
// ---------------------------------------------------------------------------

typedef __attribute__((ext_vector_type(8)))  __bf16 v8bf;
typedef __attribute__((ext_vector_type(16))) __bf16 v16bf;
typedef __attribute__((ext_vector_type(8)))  float  v8f;

#define WMMA_BF16(a, b, c) \
  __builtin_amdgcn_wmma_f32_16x16x32_bf16(false, (a), false, (b), (short)0, (c), false, false)

static constexpr int Bn  = 4;
static constexpr int Cch = 64;
static constexpr int Hd  = 192;
static constexpr int Wd  = 192;
static constexpr int CHW = Hd * Wd;     // 36864 (f32 NCHW plane stride)
static constexpr int KW  = 576;         // conv GEMM K = 64*9

__device__ __forceinline__ unsigned short f2bf(float f) {
  union { float f; unsigned int u; } cv; cv.f = f;
  unsigned int u = cv.u;
  unsigned int r = u + 0x7FFFu + ((u >> 16) & 1u);   // round-to-nearest-even
  return (unsigned short)(r >> 16);
}
__device__ __forceinline__ float bf2f(unsigned short h) {
  union { unsigned int u; float f; } cv; cv.u = ((unsigned int)h) << 16;
  return cv.f;
}

// ---- CDNA5 async LDS staging (ASYNCcnt path) ------------------------------
__device__ __forceinline__ unsigned int lds_addr(const void* p) {
  // generic LDS pointer: addr[31:0] is the LDS byte offset (ISA aperture rule)
  return (unsigned int)(uintptr_t)p;
}
__device__ __forceinline__ void async_b128(unsigned int lds_off, const void* gptr) {
  unsigned long long ga = (unsigned long long)(uintptr_t)gptr;
  asm volatile("global_load_async_to_lds_b128 %0, %1, off"
               :: "v"(lds_off), "v"(ga) : "memory");
}
__device__ __forceinline__ void async_wait0() {
  asm volatile("s_wait_asynccnt 0" ::: "memory");
}

// ---- WMMA fragment loads ---------------------------------------------------
// A (16x32 bf16): per lane 8 contiguous K + 8 contiguous K at +16.
__device__ __forceinline__ v16bf ld_fragA(const unsigned short* p) {
  v8bf lo = *(const v8bf*)(const void*)p;
  v8bf hi = *(const v8bf*)(const void*)(p + 16);
  return __builtin_shufflevector(lo, hi, 0,1,2,3,4,5,6,7,8,9,10,11,12,13,14,15);
}
// B (32x16 bf16): per lane 8 contiguous K + 8 contiguous K at +16.
__device__ __forceinline__ v16bf ld_fragB(const unsigned short* p) {
  v8bf lo = *(const v8bf*)(const void*)p;
  v8bf hi = *(const v8bf*)(const void*)(p + 16);
  return __builtin_shufflevector(lo, hi, 0,1,2,3,4,5,6,7,8,9,10,11,12,13,14,15);
}
__device__ __forceinline__ v8f vzero8() {
  v8f z;
#pragma unroll
  for (int i = 0; i < 8; ++i) z[i] = 0.0f;
  return z;
}

// ---------------------------------------------------------------------------
// f32 NCHW -> bf16 NHWC transpose/convert (tiled through LDS, both sides
// coalesced). Grid = B*H*3, block 256; tile = 64(w) x 64(c).
// ---------------------------------------------------------------------------
__global__ __launch_bounds__(256)
void nchw_to_nhwc_bf16(const float* __restrict__ src, unsigned short* __restrict__ dst) {
  __shared__ unsigned short tile[64 * 68];
  int blk = blockIdx.x;
  int b   = blk / (Hd * 3);
  int rem = blk % (Hd * 3);
  int h   = rem / 3;
  int wB  = (rem % 3) * 64;
  int tid = threadIdx.x;
  for (int idx = tid; idx < 64 * 64; idx += 256) {
    int c = idx / 64, w = idx % 64;                       // read: w contiguous
    tile[w * 68 + c] = f2bf(src[((b * Cch + c) * Hd + h) * Wd + wB + w]);
  }
  __syncthreads();
  for (int idx = tid; idx < 64 * 64; idx += 256) {
    int w = idx / 64, c = idx % 64;                       // write: c contiguous
    dst[((size_t)(b * Hd + h) * Wd + wB + w) * 64 + c] = tile[w * 68 + c];
  }
}

// ---------------------------------------------------------------------------
// Weight packing: f32 OIHW [64][64][3][3] -> bf16 [o][tap*64 + ci].
// ---------------------------------------------------------------------------
__global__ void pack_weights_kernel(const float* __restrict__ src,
                                    unsigned short* __restrict__ dst) {
  int i = blockIdx.x * blockDim.x + threadIdx.x;
  if (i >= Cch * KW) return;
  int o = i / KW, k = i % KW, tap = k / 64, ci = k % 64;
  dst[i] = f2bf(src[(o * Cch + ci) * 9 + tap]);
}

// ---------------------------------------------------------------------------
// 3x3 conv (stride 1, pad 1), bf16 NHWC in/out, implicit GEMM.
// Block = 128 (4 waves): wave = 16(w) x 64(o) tile; grid = B*H*3.
// Input halo staged with async b128 copies (zero-fill halo via ds stores).
// ---------------------------------------------------------------------------
__global__ __launch_bounds__(128)
void conv3x3_wmma(const unsigned short* __restrict__ x,     // bf16 NHWC
                  const unsigned short* __restrict__ wp,    // bf16 [64][576]
                  const float* __restrict__ bias,
                  const unsigned short* __restrict__ skip,  // bf16 NHWC or null
                  unsigned short* __restrict__ y,           // bf16 NHWC
                  int do_relu) {
  __shared__ unsigned short sIn[3 * 66 * 64];   // [row][sw][c], c-contiguous

  int blk  = blockIdx.x;
  int b    = blk / (Hd * 3);
  int rem  = blk % (Hd * 3);
  int h    = rem / 3;
  int wB   = (rem % 3) * 64;
  int tid  = threadIdx.x;

  __builtin_prefetch(wp + (tid & 63) * 64, 0, 1);   // warm packed weights

  // Stage 3 rows x 66 w-positions x 64 ch in 16B chunks: async copy in-bounds,
  // zero-fill out-of-bounds halo.
  for (int idx = tid; idx < 3 * 66 * 8; idx += 128) {
    int part = idx & 7;
    int t2   = idx >> 3;
    int sw   = t2 % 66;
    int r    = t2 / 66;
    int gh   = h + r - 1;
    int gw   = wB + sw - 1;
    unsigned short* lp = &sIn[((r * 66 + sw) * 64) + part * 8];
    if ((unsigned)gh < (unsigned)Hd && (unsigned)gw < (unsigned)Wd) {
      const unsigned short* gp = x + ((size_t)(b * Hd + gh) * Wd + gw) * 64 + part * 8;
      async_b128(lds_addr(lp), gp);
    } else {
      *(uint4*)(void*)lp = make_uint4(0u, 0u, 0u, 0u);
    }
  }
  async_wait0();
  __syncthreads();

  int wv = tid >> 5, lane = tid & 31;
  int hf = lane >> 4, nl = lane & 15;

  v8f acc[4];
#pragma unroll
  for (int i = 0; i < 4; ++i) acc[i] = vzero8();

  for (int tap = 0; tap < 9; ++tap) {
    int dh = tap / 3, dw = tap % 3;
#pragma unroll
    for (int kc = 0; kc < 2; ++kc) {
      v16bf a = ld_fragA(&sIn[(dh * 66 + wv * 16 + nl + dw) * 64 + kc * 32 + hf * 8]);
#pragma unroll
      for (int nt = 0; nt < 4; ++nt) {
        v16bf bb = ld_fragB(wp + (nt * 16 + nl) * KW + tap * 64 + kc * 32 + hf * 8);
        acc[nt] = WMMA_BF16(a, bb, acc[nt]);
      }
    }
  }

  // Epilogue: bias (+relu) (+skip), store bf16 NHWC (c-contiguous).
#pragma unroll
  for (int nt = 0; nt < 4; ++nt) {
    int o = nt * 16 + nl;
    float bv = bias[o];
#pragma unroll
    for (int r = 0; r < 8; ++r) {
      int M = r + hf * 8;
      int w = wB + wv * 16 + M;
      size_t yi = ((size_t)(b * Hd + h) * Wd + w) * 64 + o;
      float v = acc[nt][r] + bv;
      if (do_relu) v = fmaxf(v, 0.0f);
      if (skip) v += bf2f(skip[yi]);
      y[yi] = f2bf(v);
    }
  }
}

// ---------------------------------------------------------------------------
// Per-(b,h) row attention. Block = 384 (12 waves), grid = B*H.
// Q/K are bf16 NHWC: one (b,h) slab = 192*64 bf16 contiguous -> pure async
// staging. Dynamic LDS (ushort elems):
//   QT [192][64]  A for score GEMM      KT [192][64]  B for score GEMM
//   F1 [64][200]  low1 row   F2 low2 row   G1 r2l1   G2 l2r1
//   SB [192][200] S          STB [192][200] S^T
// Total = 305,152 bytes (< 320 KB WGP LDS).
// ---------------------------------------------------------------------------
__global__ __launch_bounds__(384)
void attn_rows_wmma(const float* __restrict__ low1, const float* __restrict__ low2,
                    const unsigned short* __restrict__ Q1, const unsigned short* __restrict__ K1,
                    const unsigned short* __restrict__ Q2, const unsigned short* __restrict__ K2,
                    float* __restrict__ outL, float* __restrict__ outR) {
  extern __shared__ unsigned short sm[];
  unsigned short* QT  = sm;                     // 192*64
  unsigned short* KT  = QT  + 192 * 64;
  unsigned short* F1  = KT  + 192 * 64;         // 64*200
  unsigned short* F2  = F1  + 64 * 200;
  unsigned short* G1  = F2  + 64 * 200;
  unsigned short* G2  = G1  + 64 * 200;
  unsigned short* SB  = G2  + 64 * 200;         // 192*200
  unsigned short* STB = SB  + 192 * 200;

  int b = blockIdx.x / Hd;
  int h = blockIdx.x % Hd;
  int tid  = threadIdx.x;
  int wv   = tid >> 5;       // 0..11
  int lane = tid & 31;
  int hf   = lane >> 4;
  int nl   = lane & 15;
  int w0   = wv * 16;

  int base0 = b * Cch * CHW + h * Wd;                      // f32 NCHW: + c*CHW + w
  size_t slab = (size_t)(b * Hd + h) * Wd * 64;            // bf16 NHWC row slab

  // Stage feature rows (f32 NCHW -> bf16 [c][v]).
  for (int idx = tid; idx < 64 * 192; idx += 384) {
    int c = idx / 192, w = idx % 192;
    F1[c * 200 + w] = f2bf(low1[base0 + c * CHW + w]);
    F2[c * 200 + w] = f2bf(low2[base0 + c * CHW + w]);
  }

  v8f leftAcc[4], rightAcc[4];
#pragma unroll
  for (int i = 0; i < 4; ++i) { leftAcc[i] = vzero8(); rightAcc[i] = vzero8(); }

  for (int pass = 0; pass < 2; ++pass) {
    const unsigned short* Qg = (pass ? Q2 : Q1) + slab;
    const unsigned short* Kg = (pass ? K2 : K1) + slab;

    __syncthreads();   // F staged (pass0) / previous SB,STB consumers done (pass1)

    // Async-stage Q^T / K^T slabs: 1536 x 16B chunks each, 4 per thread.
#pragma unroll
    for (int i = 0; i < 4; ++i) {
      int chunk = tid + i * 384;
      async_b128(lds_addr(&QT[chunk * 8]), Qg + chunk * 8);
      async_b128(lds_addr(&KT[chunk * 8]), Kg + chunk * 8);
    }
    async_wait0();
    __syncthreads();

    // Score GEMM: S[w][v] = sum_c Q[w][c] K[v][c]; M-tile = w0, N = 192, K = 64.
    v8f sacc[12];
#pragma unroll
    for (int t = 0; t < 12; ++t) sacc[t] = vzero8();
#pragma unroll
    for (int kc = 0; kc < 2; ++kc) {
      v16bf a = ld_fragA(&QT[(w0 + nl) * 64 + kc * 32 + hf * 8]);
#pragma unroll
      for (int t = 0; t < 12; ++t) {
        v16bf bb = ld_fragB(&KT[(t * 16 + nl) * 64 + kc * 32 + hf * 8]);
        sacc[t] = WMMA_BF16(a, bb, sacc[t]);
      }
    }

    // Row softmax; row M = r + hf*8 spans 12 regs x 16 lanes of this half.
#pragma unroll
    for (int r = 0; r < 8; ++r) {
      float m = -3.0e38f;
#pragma unroll
      for (int t = 0; t < 12; ++t) m = fmaxf(m, sacc[t][r]);
      m = fmaxf(m, __shfl_xor(m, 1, 32));
      m = fmaxf(m, __shfl_xor(m, 2, 32));
      m = fmaxf(m, __shfl_xor(m, 4, 32));
      m = fmaxf(m, __shfl_xor(m, 8, 32));
      float s = 0.0f;
#pragma unroll
      for (int t = 0; t < 12; ++t) {
        float e = __expf(sacc[t][r] - m);
        sacc[t][r] = e;
        s += e;
      }
      s += __shfl_xor(s, 1, 32);
      s += __shfl_xor(s, 2, 32);
      s += __shfl_xor(s, 4, 32);
      s += __shfl_xor(s, 8, 32);
      float inv = 1.0f / s;
      int wrow = w0 + r + hf * 8;
#pragma unroll
      for (int t = 0; t < 12; ++t) {
        float v = fmaxf(sacc[t][r] * inv, 0.0f);     // relu(softmax) == softmax
        unsigned short u = f2bf(v);
        int vcol = t * 16 + nl;
        SB[wrow * 200 + vcol]  = u;    // S
        STB[vcol * 200 + wrow] = u;    // S^T
      }
    }
    __syncthreads();

    if (pass == 0) {
      // r2l1 = apply(S1, low2) -> left, G1 ; l2r1 = apply_T(S1, low1) -> right, G2
      v8f t1[4], t2[4];
#pragma unroll
      for (int i = 0; i < 4; ++i) { t1[i] = vzero8(); t2[i] = vzero8(); }
      for (int kc = 0; kc < 6; ++kc) {
        v16bf bS  = ld_fragB(&SB [(w0 + nl) * 200 + kc * 32 + hf * 8]);
        v16bf bST = ld_fragB(&STB[(w0 + nl) * 200 + kc * 32 + hf * 8]);
#pragma unroll
        for (int mt = 0; mt < 4; ++mt) {
          v16bf aF2 = ld_fragA(&F2[(mt * 16 + nl) * 200 + kc * 32 + hf * 8]);
          t1[mt] = WMMA_BF16(aF2, bS, t1[mt]);
          v16bf aF1 = ld_fragA(&F1[(mt * 16 + nl) * 200 + kc * 32 + hf * 8]);
          t2[mt] = WMMA_BF16(aF1, bST, t2[mt]);
        }
      }
#pragma unroll
      for (int mt = 0; mt < 4; ++mt) {
        leftAcc[mt]  += t1[mt];
        rightAcc[mt] += t2[mt];
#pragma unroll
        for (int r = 0; r < 8; ++r) {
          int c = mt * 16 + r + hf * 8;
          int w = w0 + nl;
          G1[c * 200 + w] = f2bf(t1[mt][r]);
          G2[c * 200 + w] = f2bf(t2[mt][r]);
        }
      }
    } else {
      // r2l2 (F2*S2^T) + l2r2l (G2*S2^T) -> left ; l2r2 (F1*S2) + r2l2r (G1*S2) -> right
      for (int kc = 0; kc < 6; ++kc) {
        v16bf bS  = ld_fragB(&SB [(w0 + nl) * 200 + kc * 32 + hf * 8]);
        v16bf bST = ld_fragB(&STB[(w0 + nl) * 200 + kc * 32 + hf * 8]);
#pragma unroll
        for (int mt = 0; mt < 4; ++mt) {
          v16bf aF2 = ld_fragA(&F2[(mt * 16 + nl) * 200 + kc * 32 + hf * 8]);
          leftAcc[mt] = WMMA_BF16(aF2, bS, leftAcc[mt]);
          v16bf aG2 = ld_fragA(&G2[(mt * 16 + nl) * 200 + kc * 32 + hf * 8]);
          leftAcc[mt] = WMMA_BF16(aG2, bS, leftAcc[mt]);
          v16bf aF1 = ld_fragA(&F1[(mt * 16 + nl) * 200 + kc * 32 + hf * 8]);
          rightAcc[mt] = WMMA_BF16(aF1, bST, rightAcc[mt]);
          v16bf aG1 = ld_fragA(&G1[(mt * 16 + nl) * 200 + kc * 32 + hf * 8]);
          rightAcc[mt] = WMMA_BF16(aG1, bST, rightAcc[mt]);
        }
      }
    }
  }

  // Final: left = low1 + ..., right = low2 + ...  (f32 NCHW outputs)
#pragma unroll
  for (int mt = 0; mt < 4; ++mt) {
#pragma unroll
    for (int r = 0; r < 8; ++r) {
      int c = mt * 16 + r + hf * 8;
      int w = w0 + nl;
      int idx = (b * Cch + c) * CHW + h * Wd + w;
      outL[idx] = low1[idx] + leftAcc[mt][r];
      outR[idx] = low2[idx] + rightAcc[mt][r];
    }
  }
}

// ---------------------------------------------------------------------------
// Host launcher.
// ---------------------------------------------------------------------------
extern "C" void kernel_launch(void* const* d_in, const int* in_sizes, int n_in,
                              void* d_out, int out_size, void* d_ws, size_t ws_size,
                              hipStream_t stream) {
  (void)in_sizes; (void)n_in; (void)out_size; (void)ws_size;

  const float* low1   = (const float*)d_in[0];
  const float* low2   = (const float*)d_in[1];
  const float* fe1_w1 = (const float*)d_in[2];
  const float* fe1_b1 = (const float*)d_in[3];
  const float* fe1_w2 = (const float*)d_in[4];
  const float* fe1_b2 = (const float*)d_in[5];
  const float* fe2_w1 = (const float*)d_in[6];
  const float* fe2_b1 = (const float*)d_in[7];
  const float* fe2_w2 = (const float*)d_in[8];
  const float* fe2_b2 = (const float*)d_in[9];
  const float* conv_w = (const float*)d_in[10];
  const float* conv_b = (const float*)d_in[11];

  float* out = (float*)d_out;
  const size_t NBUF = (size_t)Bn * Cch * Hd * Wd;     // 9,437,184 elems
  float* outL = out;
  float* outR = out + NBUF;

  // Workspace: 8 bf16 NHWC activation buffers + 5 packed bf16 weight blocks.
  unsigned short* ws   = (unsigned short*)d_ws;
  unsigned short* loT1 = ws + 0 * NBUF;   // low1 bf16 NHWC
  unsigned short* loT2 = ws + 1 * NBUF;   // low2 bf16 NHWC
  unsigned short* tmpb = ws + 2 * NBUF;
  unsigned short* csmb = ws + 3 * NBUF;
  unsigned short* Q1   = ws + 4 * NBUF;
  unsigned short* K1   = ws + 5 * NBUF;
  unsigned short* Q2   = ws + 6 * NBUF;
  unsigned short* K2   = ws + 7 * NBUF;
  unsigned short* wpk  = ws + 8 * NBUF;
  const int WPN = Cch * KW;                            // 36,864 elems
  unsigned short* wp_fe1w1 = wpk + 0 * WPN;
  unsigned short* wp_fe1w2 = wpk + 1 * WPN;
  unsigned short* wp_fe2w1 = wpk + 2 * WPN;
  unsigned short* wp_fe2w2 = wpk + 3 * WPN;
  unsigned short* wp_conv  = wpk + 4 * WPN;

  dim3 pgrid((WPN + 255) / 256), pblk(256);
  pack_weights_kernel<<<pgrid, pblk, 0, stream>>>(fe1_w1, wp_fe1w1);
  pack_weights_kernel<<<pgrid, pblk, 0, stream>>>(fe1_w2, wp_fe1w2);
  pack_weights_kernel<<<pgrid, pblk, 0, stream>>>(fe2_w1, wp_fe2w1);
  pack_weights_kernel<<<pgrid, pblk, 0, stream>>>(fe2_w2, wp_fe2w2);
  pack_weights_kernel<<<pgrid, pblk, 0, stream>>>(conv_w, wp_conv);

  dim3 tgrid(Bn * Hd * 3), tblk(256);
  nchw_to_nhwc_bf16<<<tgrid, tblk, 0, stream>>>(low1, loT1);
  nchw_to_nhwc_bf16<<<tgrid, tblk, 0, stream>>>(low2, loT2);

  dim3 cgrid(Bn * Hd * 3), cblk(128);
  // Q1 = conv(csm(low1, fe1))
  conv3x3_wmma<<<cgrid, cblk, 0, stream>>>(loT1, wp_fe1w1, fe1_b1, nullptr, tmpb, 1);
  conv3x3_wmma<<<cgrid, cblk, 0, stream>>>(tmpb, wp_fe1w2, fe1_b2, loT1,    csmb, 0);
  conv3x3_wmma<<<cgrid, cblk, 0, stream>>>(csmb, wp_conv,  conv_b, nullptr, Q1,   0);
  // K1 = conv(csm(low2, fe1))
  conv3x3_wmma<<<cgrid, cblk, 0, stream>>>(loT2, wp_fe1w1, fe1_b1, nullptr, tmpb, 1);
  conv3x3_wmma<<<cgrid, cblk, 0, stream>>>(tmpb, wp_fe1w2, fe1_b2, loT2,    csmb, 0);
  conv3x3_wmma<<<cgrid, cblk, 0, stream>>>(csmb, wp_conv,  conv_b, nullptr, K1,   0);
  // Q2 = conv(csm(low1, fe2))
  conv3x3_wmma<<<cgrid, cblk, 0, stream>>>(loT1, wp_fe2w1, fe2_b1, nullptr, tmpb, 1);
  conv3x3_wmma<<<cgrid, cblk, 0, stream>>>(tmpb, wp_fe2w2, fe2_b2, loT1,    csmb, 0);
  conv3x3_wmma<<<cgrid, cblk, 0, stream>>>(csmb, wp_conv,  conv_b, nullptr, Q2,   0);
  // K2 = conv(csm(low2, fe2))
  conv3x3_wmma<<<cgrid, cblk, 0, stream>>>(loT2, wp_fe2w1, fe2_b1, nullptr, tmpb, 1);
  conv3x3_wmma<<<cgrid, cblk, 0, stream>>>(tmpb, wp_fe2w2, fe2_b2, loT2,    csmb, 0);
  conv3x3_wmma<<<cgrid, cblk, 0, stream>>>(csmb, wp_conv,  conv_b, nullptr, K2,   0);

  // Row attention + output fusion.
  const unsigned int smemBytes =
      (192 * 64 * 2 + 64 * 200 * 4 + 192 * 200 * 2) * sizeof(unsigned short); // 305,152
  dim3 agrid(Bn * Hd), ablk(384);
  attn_rows_wmma<<<agrid, ablk, smemBytes, stream>>>(low1, low2, Q1, K1, Q2, K2,
                                                     outL, outR);
}
